// MultiHeadAttention_833223655722
// MI455X (gfx1250) — matrix operand order repaired and checked by reading, not compiled, not run on hardware
//
#include <hip/hip_runtime.h>
#include <hip/hip_bf16.h>
#include <math.h>

// Problem dims
#define BB  16
#define SS  512
#define EE  4096
#define HH  8
#define HD_ 512
#define BSR (BB*SS)      // 8192 token rows

typedef __attribute__((ext_vector_type(16))) __bf16 bf16x16;
typedef __attribute__((ext_vector_type(8)))  float  f32x8;

union FragU { bf16x16 v; uint4 q[2]; };

#ifndef __has_builtin
#define __has_builtin(x) 0
#endif
#if __has_builtin(__builtin_amdgcn_global_load_async_to_lds_b128)
#define HAVE_ASYNC_LDS 1
#else
#define HAVE_ASYNC_LDS 0
#endif

#if HAVE_ASYNC_LDS
// Builtin signature (from hipcc diagnostic): param 1 is
// 'int __attribute__((vector_size(16))) __device__*' (AS1); param 2 LDS (AS3).
typedef int v4i_t __attribute__((vector_size(16)));
typedef __attribute__((address_space(1))) v4i_t* as1_v4i;
typedef __attribute__((address_space(3))) v4i_t* as3_v4i;
#endif

// fp32 -> bf16 via native cast (backend selects v_cvt_*bf16_f32)
__device__ __forceinline__ unsigned short f2bf(float f) {
  union { __bf16 b; unsigned short s; } c; c.b = (__bf16)f; return c.s;
}
__device__ __forceinline__ float bf2f(unsigned short h) {
  union { unsigned int u; float f; } c; c.u = ((unsigned int)h) << 16;
  return c.f;
}
__device__ __forceinline__ ushort4 pack4(float4 v) {
  return make_ushort4(f2bf(v.x), f2bf(v.y), f2bf(v.z), f2bf(v.w));
}
__device__ __forceinline__ float gelu_f(float x) {
  return 0.5f * x * (1.0f + erff(x * 0.70710678118654752f));
}

// ---- global->LDS 32B copy: async (ASYNCcnt) when available ----------------
__device__ __forceinline__ void g2l_b128x2(unsigned short* dst,
                                           const unsigned short* src) {
#if HAVE_ASYNC_LDS
  as1_v4i g = (as1_v4i)(unsigned long long)(const void*)src;
  as3_v4i l = (as3_v4i)(unsigned int)(unsigned long long)(void*)dst;
  __builtin_amdgcn_global_load_async_to_lds_b128(g, l, 0,  0);
  __builtin_amdgcn_global_load_async_to_lds_b128(g, l, 16, 0);
#else
  uint4 v0 = *(const uint4*)(src);
  uint4 v1 = *(const uint4*)(src + 8);
  *(uint4*)(dst)     = v0;
  *(uint4*)(dst + 8) = v1;
#endif
}
__device__ __forceinline__ void g2l_commit() {
#if HAVE_ASYNC_LDS
#if __has_builtin(__builtin_amdgcn_s_wait_asynccnt)
  __builtin_amdgcn_s_wait_asynccnt(0);
#else
  asm volatile("s_wait_asynccnt 0x0" ::: "memory");
#endif
#endif
}

// ---- WMMA fragment loaders (per CDNA5 ISA 7.12.2 layouts) -----------------
// A: 16x32 bf16, lane m=l&15; K chunks {0..7,16..23} (l<16) / {8..15,24..31}
__device__ __forceinline__ bf16x16 load_a_frag(const unsigned short* lds,
                                               int row0, int k0, int stride) {
  const int lane = threadIdx.x & 31;
  const int m  = row0 + (lane & 15);
  const int kk = k0 + ((lane >> 4) << 3);
  FragU a;
  a.q[0] = *(const uint4*)(lds + m * stride + kk);
  a.q[1] = *(const uint4*)(lds + m * stride + kk + 16);
  return a.v;
}
// B: 32x16 (KxN), operand stored [N][K] row-major; lane n=l&15 holds K chunks
// {0..7,16..23} (l<16) or {8..15,24..31} (l>=16) -> two b128 loads.
__device__ __forceinline__ bf16x16 load_b_frag(const unsigned short* lds,
                                               int n0, int k0, int stride) {
  const int lane = threadIdx.x & 31;
  const int n  = n0 + (lane & 15);
  const int kk = k0 + ((lane >> 4) << 3);
  FragU b;
  b.q[0] = *(const uint4*)(lds + n * stride + kk);
  b.q[1] = *(const uint4*)(lds + n * stride + kk + 16);
  return b.v;
}
__device__ __forceinline__ f32x8 wmma_bf16(bf16x16 a, bf16x16 b, f32x8 c) {
  return __builtin_amdgcn_wmma_f32_16x16x32_bf16(false, a, false, b,
                                                 (short)0, c, false, false);
}

// ---- fp32 -> bf16 bulk convert --------------------------------------------
__global__ __launch_bounds__(256)
void cvt_f32_bf16(const float* __restrict__ src, unsigned short* __restrict__ dst,
                  int n4) {
  int i = blockIdx.x * blockDim.x + threadIdx.x;
  int stride = gridDim.x * blockDim.x;
  for (; i < n4; i += stride) {
    float4 v = ((const float4*)src)[i];
    ((ushort4*)dst)[i] = pack4(v);
  }
}

// ---- big GEMM: C[M=8192,N=4096] = A(bf16)[M,K=4096] * W(fp32)[N,K]^T + b --
// 128x128 workgroup tile, 8 waves (4 along M x 2 along N), wave tile 32x64:
// 8 WMMA per k-step fed by 12 ds_load_b128 (1.5 DS/WMMA).
#define GM   128
#define GN   128
#define LSTR 48      // padded LDS row stride (elems); 96B rows, 16B-aligned

template<int MODE>   // 0=Q, 1=K (B,H,S,HD bf16); 2=V transposed (B,H,HD,S); 3=final fp32
__global__ __launch_bounds__(256)
void gemm_proj(const unsigned short* __restrict__ A,
               const float* __restrict__ W,
               const float* __restrict__ bias,
               void* __restrict__ outp) {
  __shared__ __align__(16) unsigned short As[GM * LSTR];
  __shared__ __align__(16) unsigned short Bs[GN * LSTR];

  const int tid   = threadIdx.x;
  const int lane  = tid & 31;
  const int wave  = tid >> 5;
  const int waveM = wave >> 1;           // 0..3  (32 rows each)
  const int waveN = wave & 1;            // 0..1  (64 cols each)
  const int m0 = blockIdx.x * GM;
  const int n0 = blockIdx.y * GN;

  const f32x8 z8 = {0.f,0.f,0.f,0.f,0.f,0.f,0.f,0.f};
  f32x8 acc[2][4] = {{z8, z8, z8, z8}, {z8, z8, z8, z8}};

  const int arow = tid >> 1;             // 0..127
  const int acol = (tid & 1) * 16;       // 16 bf16 per thread (32B)
  const int brow = tid >> 1;             // 0..127
  const int bcol = (tid & 1) * 16;       // 16 fp32 per thread

  for (int k0 = 0; k0 < EE; k0 += 32) {
    // W (fp32) staged through VGPRs for conversion; overlaps prior compute
    const float* wsrc = W + (size_t)(n0 + brow) * EE + k0 + bcol;
    float4 f0 = ((const float4*)wsrc)[0];
    float4 f1 = ((const float4*)wsrc)[1];
    float4 f2 = ((const float4*)wsrc)[2];
    float4 f3 = ((const float4*)wsrc)[3];
    if (k0 + 32 < EE) __builtin_prefetch((const void*)(wsrc + 32), 0, 1);
    __syncthreads();                                   // prev compute done
    // A (bf16): direct async global->LDS copy
    g2l_b128x2(As + arow * LSTR + acol,
               A + (size_t)(m0 + arow) * EE + k0 + acol);
    ushort4* bd = (ushort4*)(Bs + brow * LSTR + bcol);
    bd[0] = pack4(f0); bd[1] = pack4(f1); bd[2] = pack4(f2); bd[3] = pack4(f3);
    g2l_commit();                                      // drain ASYNCcnt
    __syncthreads();                                   // tiles visible
    bf16x16 a0 = load_a_frag(As, waveM * 32,      0, LSTR);
    bf16x16 a1 = load_a_frag(As, waveM * 32 + 16, 0, LSTR);
#pragma unroll
    for (int wn = 0; wn < 4; ++wn) {
      bf16x16 b = load_b_frag(Bs, waveN * 64 + wn * 16, 0, LSTR);
      acc[0][wn] = wmma_bf16(a0, b, acc[0][wn]);
      acc[1][wn] = wmma_bf16(a1, b, acc[1][wn]);
    }
  }

#pragma unroll
  for (int wm = 0; wm < 2; ++wm)
#pragma unroll
    for (int wn = 0; wn < 4; ++wn) {
      const int mb = m0 + waveM * 32 + wm * 16 + ((lane >> 4) << 3);
      const int nn = n0 + waveN * 64 + wn * 16 + (lane & 15);
      const float bv = bias[nn];
#pragma unroll
      for (int r = 0; r < 8; ++r) {
        const int m = mb + r;
        const float val = acc[wm][wn][r] + bv;
        if (MODE == 3) {
          ((float*)outp)[(size_t)m * EE + nn] = val;
        } else {
          const int b = m >> 9, s = m & 511;
          const int h = nn >> 9, d = nn & 511;
          unsigned short* o = (unsigned short*)outp;
          if (MODE == 2)  // v stored transposed: (B,H,HD,S)
            o[(((size_t)b * HH + h) * HD_ + d) * SS + s] = f2bf(val);
          else            // q,k: (B,H,S,HD)
            o[(((size_t)b * HH + h) * SS + s) * HD_ + d] = f2bf(val);
        }
      }
    }
}

// ---- LayerNorm over HD=512, one wave32 per row, in-place on bf16 ----------
__global__ __launch_bounds__(256)
void ln_kernel(unsigned short* __restrict__ t,
               const float* __restrict__ g, const float* __restrict__ b) {
  const int row  = blockIdx.x * 8 + (threadIdx.x >> 5);
  const int lane = threadIdx.x & 31;
  unsigned short* p = t + (size_t)row * HD_ + lane * 16;
  union { uint4 q[2]; unsigned short s[16]; } u;
  u.q[0] = *(const uint4*)(p);
  u.q[1] = *(const uint4*)(p + 8);
  float x[16], sum = 0.f, sq = 0.f;
#pragma unroll
  for (int i = 0; i < 16; ++i) { x[i] = bf2f(u.s[i]); sum += x[i]; sq += x[i]*x[i]; }
#pragma unroll
  for (int m = 16; m >= 1; m >>= 1) {
    sum += __shfl_xor(sum, m, 32);
    sq  += __shfl_xor(sq,  m, 32);
  }
  const float mu = sum * (1.0f / HD_);
  const float rs = rsqrtf(sq * (1.0f / HD_) - mu * mu + 1e-5f);
#pragma unroll
  for (int i = 0; i < 16; ++i) {
    const int d = lane * 16 + i;
    u.s[i] = f2bf((x[i] - mu) * rs * g[d] + b[d]);
  }
  *(uint4*)(p)     = u.q[0];
  *(uint4*)(p + 8) = u.q[1];
}

// ---- fused attention core helpers -----------------------------------------
#define ASTR 520   // padded stride for 32x512 LDS matrices (1040B: 16B-aligned)

__device__ __forceinline__ void tile_gemm_bf16src(
    const unsigned short* __restrict__ gsrc,   // 128-row chunk, [row][K=512]
    const unsigned short* smAlds, int a_row0,
    unsigned short* smT, f32x8& acc0, f32x8& acc1, int waveN) {
  const int rowb = threadIdx.x >> 1;
  const int colb = (threadIdx.x & 1) * 16;
  for (int k0 = 0; k0 < 512; k0 += 32) {
    __syncthreads();                           // prev frag reads done
    g2l_b128x2(smT + rowb * LSTR + colb,
               gsrc + (size_t)rowb * 512 + k0 + colb);
    g2l_commit();
    __syncthreads();
    bf16x16 a  = load_a_frag(smAlds, a_row0, k0, ASTR);
    bf16x16 b0 = load_b_frag(smT, waveN * 32,      0, LSTR);
    bf16x16 b1 = load_b_frag(smT, waveN * 32 + 16, 0, LSTR);
    acc0 = wmma_bf16(a, b0, acc0);
    acc1 = wmma_bf16(a, b1, acc1);
  }
}

__device__ __forceinline__ void tile_gemm_f32src(
    const float* __restrict__ gsrc,            // 128-row chunk, [row][K=512] fp32
    const unsigned short* smAlds, int a_row0,
    unsigned short* smT, f32x8& acc0, f32x8& acc1, int waveN) {
  const int rowb = threadIdx.x >> 1;
  const int colb = (threadIdx.x & 1) * 16;
  for (int k0 = 0; k0 < 512; k0 += 32) {
    const float* src = gsrc + (size_t)rowb * 512 + k0 + colb;
    float4 f0 = ((const float4*)src)[0];
    float4 f1 = ((const float4*)src)[1];
    float4 f2 = ((const float4*)src)[2];
    float4 f3 = ((const float4*)src)[3];
    __syncthreads();
    ushort4* d = (ushort4*)(smT + rowb * LSTR + colb);
    d[0] = pack4(f0); d[1] = pack4(f1); d[2] = pack4(f2); d[3] = pack4(f3);
    __syncthreads();
    bf16x16 a  = load_a_frag(smAlds, a_row0, k0, ASTR);
    bf16x16 b0 = load_b_frag(smT, waveN * 32,      0, LSTR);
    bf16x16 b1 = load_b_frag(smT, waveN * 32 + 16, 0, LSTR);
    acc0 = wmma_bf16(a, b0, acc0);
    acc1 = wmma_bf16(a, b1, acc1);
  }
}

// ---- fused: scores -> (gate|val)@Wg -> GeLU gate -> L2 norm -> @v ---------
// One workgroup = one (b,h) and 32 rows of s. Scores never touch HBM
// (saves ~1 GB of round-trip traffic vs materializing B,H,S,S fp32).
__global__ __launch_bounds__(256)
void attn_fused(const unsigned short* __restrict__ qb,
                const unsigned short* __restrict__ kb,
                const unsigned short* __restrict__ vtb,
                const float* __restrict__ Wg,
                const float* __restrict__ bgp,
                unsigned short* __restrict__ ab) {
  __shared__ __align__(16) unsigned short smA[32 * ASTR]; // q -> gelu(gate) -> w(bf16)
  __shared__ __align__(16) unsigned short smS[32 * ASTR]; // scores bf16 -> w raw
  __shared__ __align__(16) unsigned short smT[128 * LSTR];
  __shared__ float smRow[32];
  __shared__ float smR[32];

  const int tid  = threadIdx.x;
  const int lane = tid & 31;
  const int wave = tid >> 5;
  const int waveM = wave >> 2;      // 0..1 -> 16 rows each
  const int waveN = wave & 3;       // 0..3 -> 32 cols each
  const int a_row0 = waveM * 16;

  const int sb = blockIdx.x;        // s block (16)
  const int hh = blockIdx.y;        // head   (8)
  const int bb = blockIdx.z;        // batch  (16)
  const size_t bh = (size_t)bb * HH + hh;
  const int s0 = sb * 32;
  const unsigned short* qrow = qb  + (bh * SS + s0) * HD_;
  const unsigned short* krow = kb  + bh * SS * HD_;
  const unsigned short* vrow = vtb + bh * HD_ * SS;

  const f32x8 z8 = {0.f,0.f,0.f,0.f,0.f,0.f,0.f,0.f};

  // phase 0: q block (32x512 bf16) -> smA (async; drained by phase-1 commit)
  {
    const int row = tid >> 3;
    const int cs  = (tid & 7) * 64;
#pragma unroll
    for (int i = 0; i < 4; ++i)
      g2l_b128x2(smA + row * ASTR + cs + i * 16,
                 qrow + (size_t)row * HD_ + cs + i * 16);
  }

  // phase 1: scores = q k^T / sqrt(HD) -> smS (bf16)
  const float scale = 0.044194173824159216f;   // 1/sqrt(512)
  for (int nt = 0; nt < 4; ++nt) {
    f32x8 acc0 = z8, acc1 = z8;
    tile_gemm_bf16src(krow + (size_t)nt * 128 * HD_, smA, a_row0, smT, acc0, acc1, waveN);
    const int mb = a_row0 + ((lane >> 4) << 3);
    const int nbase = nt * 128 + waveN * 32 + (lane & 15);
#pragma unroll
    for (int r = 0; r < 8; ++r) {
      smS[(mb + r) * ASTR + nbase]      = f2bf(acc0[r] * scale);
      smS[(mb + r) * ASTR + nbase + 16] = f2bf(acc1[r] * scale);
    }
  }

  // phase 2: gate = scores @ Wg[512:1024]^T + bg; gelu -> smA (q is dead)
  for (int c = 0; c < 4; ++c) {
    const int nf0 = 512 + c * 128;
    f32x8 acc0 = z8, acc1 = z8;
    tile_gemm_f32src(Wg + (size_t)nf0 * 512, smS, a_row0, smT, acc0, acc1, waveN);
    const int mb = a_row0 + ((lane >> 4) << 3);
    const int nf = nf0 + waveN * 32 + (lane & 15);
    const float b0 = bgp[nf], b1 = bgp[nf + 16];
#pragma unroll
    for (int r = 0; r < 8; ++r) {
      smA[(mb + r) * ASTR + (nf - 512)]      = f2bf(gelu_f(acc0[r] + b0));
      smA[(mb + r) * ASTR + (nf - 512 + 16)] = f2bf(gelu_f(acc1[r] + b1));
    }
  }

  if (tid < 32) smRow[tid] = 0.0f;   // visible before first atomic (syncs in phase 3)

  // phase 3: val = scores @ Wg[0:512]^T + bg; w = val*gelu(gate); sumsq via ds_add_f32
  for (int c = 0; c < 4; ++c) {
    const int nf0 = c * 128;
    f32x8 acc0 = z8, acc1 = z8;
    tile_gemm_f32src(Wg + (size_t)nf0 * 512, smS, a_row0, smT, acc0, acc1, waveN);
    const int mb = a_row0 + ((lane >> 4) << 3);
    const int nf = nf0 + waveN * 32 + (lane & 15);
    const float b0 = bgp[nf], b1 = bgp[nf + 16];
#pragma unroll
    for (int r = 0; r < 8; ++r) {
      const int m = mb + r;
      const float w0 = (acc0[r] + b0) * bf2f(smA[m * ASTR + nf]);
      const float w1 = (acc1[r] + b1) * bf2f(smA[m * ASTR + nf + 16]);
      atomicAdd(&smRow[m], w0 * w0 + w1 * w1);
      smS[m * ASTR + nf]      = f2bf(w0);   // w (unnormalized) overwrites scores
      smS[m * ASTR + nf + 16] = f2bf(w1);
    }
  }

  // phase 4: L2-normalize rows of w; normalized bf16 w -> smA
  __syncthreads();
  if (tid < 32) smR[tid] = 1.0f / fmaxf(sqrtf(smRow[tid]), 1e-12f);
  __syncthreads();
  {
    const int row = tid >> 3;
    const int seg = (tid & 7) * 64;
    const float inv = smR[row];
#pragma unroll 8
    for (int i = 0; i < 64; ++i)
      smA[row * ASTR + seg + i] = f2bf(bf2f(smS[row * ASTR + seg + i]) * inv);
  }
  __syncthreads();

  // phase 5: out = w @ v  (B operand = v^T rows, contiguous), -> ab (B,S,E) bf16
  for (int c = 0; c < 4; ++c) {
    const int nd0 = c * 128;
    f32x8 acc0 = z8, acc1 = z8;
    tile_gemm_bf16src(vrow + (size_t)nd0 * SS, smA, a_row0, smT, acc0, acc1, waveN);
    const int mb = a_row0 + ((lane >> 4) << 3);
    const int nd = nd0 + waveN * 32 + (lane & 15);
#pragma unroll
    for (int r = 0; r < 8; ++r) {
      const int s = s0 + mb + r;
      unsigned short* o = ab + ((size_t)bb * SS + s) * EE + hh * HD_;
      o[nd]      = f2bf(acc0[r]);
      o[nd + 16] = f2bf(acc1[r]);
    }
  }
}

// ---------------------------------------------------------------------------
extern "C" void kernel_launch(void* const* d_in, const int* in_sizes, int n_in,
                              void* d_out, int out_size, void* d_ws, size_t ws_size,
                              hipStream_t stream) {
  (void)in_sizes; (void)n_in; (void)out_size; (void)ws_size;
  const float* x   = (const float*)d_in[0];
  const float* Wq  = (const float*)d_in[1];
  const float* bq  = (const float*)d_in[2];
  const float* Wk  = (const float*)d_in[3];
  const float* bk  = (const float*)d_in[4];
  const float* Wv  = (const float*)d_in[5];
  const float* bv  = (const float*)d_in[6];
  const float* g_q = (const float*)d_in[7];
  const float* b_q = (const float*)d_in[8];
  const float* g_k = (const float*)d_in[9];
  const float* b_k = (const float*)d_in[10];
  const float* Wg  = (const float*)d_in[11];
  const float* bg  = (const float*)d_in[12];
  const float* Wo  = (const float*)d_in[13];
  const float* bo  = (const float*)d_in[14];
  float* out = (float*)d_out;

  // Workspace: 5 bf16 tensors of B*S*E elems (~336 MB total)
  const size_t NTOK = (size_t)BSR * EE;
  unsigned short* xb  = (unsigned short*)d_ws;
  unsigned short* qb  = xb  + NTOK;
  unsigned short* kb  = qb  + NTOK;
  unsigned short* vtb = kb  + NTOK;
  unsigned short* ab  = vtb + NTOK;

  cvt_f32_bf16<<<2048, 256, 0, stream>>>(x, xb, (int)(NTOK / 4));

  dim3 gg(BSR / GM, EE / GN);            // (64, 32)
  gemm_proj<0><<<gg, 256, 0, stream>>>(xb, Wq, bq, qb);
  gemm_proj<1><<<gg, 256, 0, stream>>>(xb, Wk, bk, kb);
  gemm_proj<2><<<gg, 256, 0, stream>>>(xb, Wv, bv, vtb);

  ln_kernel<<<(BB * HH * SS) / 8, 256, 0, stream>>>(qb, g_q, b_q);
  ln_kernel<<<(BB * HH * SS) / 8, 256, 0, stream>>>(kb, g_k, b_k);

  dim3 ga(SS / 32, HH, BB);              // (16, 8, 16)
  attn_fused<<<ga, 256, 0, stream>>>(qb, kb, vtb, Wg, bg, ab);

  gemm_proj<3><<<gg, 256, 0, stream>>>(ab, Wo, bo, out);
}
